// STSA_27092653703318
// MI455X (gfx1250) — matrix operand order repaired and checked by reading, not compiled, not run on hardware
//
#include <hip/hip_runtime.h>

typedef __attribute__((ext_vector_type(16))) _Float16 v16h;
typedef __attribute__((ext_vector_type(8)))  float    v8f;

#define T_   4
#define B_   8
#define C_   256
#define N_   256
#define TN_  1024
#define DH_  32
#define HEADS_ 8

// A-matrix (16x32 f16) K index for half-slot e (0..15) and lane-half hi (0/1).
// Per ISA: lanes 0-15 hold K=0..7 then 16..23; lanes 16-31 hold K=8..15 then 24..31.
__device__ __forceinline__ int kmap16(int e, int hi) {
    int j = e >> 1, r = e & 1;
    return 2 * j + r + 8 * hi + ((j >= 4) ? 8 : 0);
}

// ---------------- f32 -> f16 convert ----------------
__global__ void cvt_f16_kernel(const float* __restrict__ src, _Float16* __restrict__ dst, int n) {
    int i = blockIdx.x * blockDim.x + threadIdx.x;
    if (i < n) dst[i] = (_Float16)src[i];
}

// ---------------- QKV GEMM (768x256 x 256x1024 per b) + BN ----------------
// Each wave: 16(M) x 64(N) tile, 4 accumulators, A-frag reused 4x per K-step.
__global__ void qkv_gemm_bn_kernel(const _Float16* __restrict__ wq, const _Float16* __restrict__ xh,
                                   const float* __restrict__ g, const float* __restrict__ be,
                                   const float* __restrict__ mu, const float* __restrict__ vr,
                                   float* __restrict__ outp) {
    int wave = (blockIdx.x * blockDim.x + threadIdx.x) >> 5;
    int lane = threadIdx.x & 31;
    int b   = wave / (48 * 16);
    int rem = wave % (48 * 16);
    int mt = rem / 16, nt = rem % 16;
    int l15 = lane & 15, hi = lane >> 4;

    v8f acc[4] = {};
    int mrow = mt * 16 + l15;
    const _Float16* xb = xh + b * (C_ * TN_) + l15 + nt * 64;
#pragma unroll
    for (int kc = 0; kc < 8; ++kc) {
        int K0 = kc * 32;
        v16h a;
#pragma unroll
        for (int e = 0; e < 16; ++e)
            a[e] = wq[mrow * 256 + K0 + kmap16(e, hi)];          // A: weights [768,256]
        v16h bf[4];
#pragma unroll
        for (int j = 0; j < 4; ++j) {
#pragma unroll
            for (int e = 0; e < 16; ++e)                          // B: lane=N col, e=K row (kmap16)
                bf[j][e] = xb[(K0 + kmap16(e, hi)) * TN_ + j * 16];
        }
#pragma unroll
        for (int j = 0; j < 4; ++j)
            acc[j] = __builtin_amdgcn_wmma_f32_16x16x32_f16(false, a, false, bf[j], (short)0, acc[j], false, false);
    }
#pragma unroll
    for (int j = 0; j < 4; ++j) {
        int ncol = nt * 64 + j * 16 + l15;
#pragma unroll
        for (int r = 0; r < 8; ++r) {
            int m = mt * 16 + r + 8 * hi;
            float y = (acc[j][r] - mu[m]) * (g[m] * rsqrtf(vr[m] + 1e-5f)) + be[m];
            outp[b * (768 * TN_) + m * TN_ + ncol] = y;
        }
    }
}

// ---------------- LIF over T on qkv, scatter spikes to head-major q/k/v (f16) ----------------
__global__ void lif_qkv_kernel(const float* __restrict__ pre,
                               _Float16* __restrict__ q, _Float16* __restrict__ k,
                               _Float16* __restrict__ v) {
    int i = blockIdx.x * blockDim.x + threadIdx.x; // B*768*N
    if (i >= B_ * 768 * N_) return;
    int n = i & 255;
    int o = (i >> 8) % 768;
    int b = i / (768 * 256);
    int part = o / 256, c = o % 256;
    int hd = c / 32, d = c % 32;
    _Float16* dst = (part == 0) ? q : ((part == 1) ? k : v);
    float mem = 0.f, sp = 0.f;
    for (int t = 0; t < T_; ++t) {
        float xv = pre[b * (768 * TN_) + o * TN_ + t * 256 + n];
        mem = (mem - sp * 0.5f) * 0.25f + xv;
        sp = rintf(fminf(fmaxf(mem, 0.f), 1.f));
        int tn = t * 256 + n;
        dst[((b * 8 + hd) * TN_ + tn) * 32 + d] = (_Float16)sp;
    }
}

// ---------------- Fused attention: s = q k^T + bias (block-causal), o = s v * 0.125 ----------------
// Each wave: 32 q-rows (two 16-row A-frags). Software-pipelined m-loop:
// k-frags double-buffered, v-frags hoisted to iteration start.
__global__ void attn_kernel(const _Float16* __restrict__ qh, const _Float16* __restrict__ kh,
                            const _Float16* __restrict__ vh, const float* __restrict__ btab,
                            float* __restrict__ opre) {
    __shared__ _Float16 slds[4][32][32];
    int wv = threadIdx.x >> 5, lane = threadIdx.x & 31;
    int bh = blockIdx.x >> 3;   // b*8+h
    int rg = blockIdx.x & 7;    // 128-row group within TN (never straddles a time block)
    int b = bh >> 3, hd = bh & 7;
    int row0 = rg * 128 + wv * 32;     // this wave's first q row
    int trow = rg >> 1;                // block-uniform time index of these rows
    int l15 = lane & 15, hi = lane >> 4;
    int base = (b * 8 + hd) * TN_ * 32;

    v16h qfA, qfB;
#pragma unroll
    for (int e = 0; e < 16; ++e) {
        int kk = kmap16(e, hi);
        qfA[e] = qh[base + (row0 + l15) * 32 + kk];
        qfB[e] = qh[base + (row0 + 16 + l15) * 32 + kk];
    }

    v8f oA0 = {}, oA1 = {}, oB0 = {}, oB1 = {};
    int nchunks = (trow + 1) * 8; // causal: m < (trow+1)*256, chunks of 32

    // prologue: k-frags for chunk 0
    v16h kf0, kf1;
#pragma unroll
    for (int e = 0; e < 16; ++e) {
        kf0[e] = kh[base + l15 * 32 + kmap16(e, hi)];        // B=k^T: lane=m col, e=dh row (kmap16)
        kf1[e] = kh[base + (16 + l15) * 32 + kmap16(e, hi)];
    }

    for (int mc = 0; mc < nchunks; ++mc) {
        int m0 = mc * 32;
        // v-frags for this chunk, issued early (consumed after LDS round-trip)
        v16h vf0, vf1;
#pragma unroll
        for (int e = 0; e < 16; ++e) {
            vf0[e] = vh[base + (m0 + kmap16(e, hi)) * 32 + l15];        // B: lane=d col, e=m row (kmap16)
            vf1[e] = vh[base + (m0 + kmap16(e, hi)) * 32 + 16 + l15];
        }
        v8f z = {};
        v8f sA0 = __builtin_amdgcn_wmma_f32_16x16x32_f16(false, qfA, false, kf0, (short)0, z, false, false);
        v8f sA1 = __builtin_amdgcn_wmma_f32_16x16x32_f16(false, qfA, false, kf1, (short)0, z, false, false);
        v8f sB0 = __builtin_amdgcn_wmma_f32_16x16x32_f16(false, qfB, false, kf0, (short)0, z, false, false);
        v8f sB1 = __builtin_amdgcn_wmma_f32_16x16x32_f16(false, qfB, false, kf1, (short)0, z, false, false);

        // double-buffer: fetch next chunk's k-frags while scores are staged/consumed
        if (mc + 1 < nchunks) {
            int m1 = m0 + 32;
#pragma unroll
            for (int e = 0; e < 16; ++e) {
                kf0[e] = kh[base + (m1 + l15) * 32 + kmap16(e, hi)];
                kf1[e] = kh[base + (m1 + 16 + l15) * 32 + kmap16(e, hi)];
            }
            __builtin_prefetch(&vh[base + (m1 + lane) * 32], 0, 1);
        }

        // add relative-position bias, stage scores to LDS for layout change (D -> A frag)
        int j0 = m0 + l15, j1 = j0 + 16;
        int tj0 = j0 >> 8, hj0 = (j0 >> 4) & 15, wj0 = j0 & 15;
        int tj1 = j1 >> 8, hj1 = (j1 >> 4) & 15, wj1 = j1 & 15;
#pragma unroll
        for (int r = 0; r < 8; ++r) {
            int qr = r + 8 * hi;
            {
                int i = row0 + qr;
                int ti = i >> 8, hy = (i >> 4) & 15, wi = i & 15;
                float bi0 = btab[(((ti - tj0 + 3) * 31 + (hy - hj0 + 15)) * 31 + (wi - wj0 + 15)) * 8 + hd];
                float bi1 = btab[(((ti - tj1 + 3) * 31 + (hy - hj1 + 15)) * 31 + (wi - wj1 + 15)) * 8 + hd];
                slds[wv][qr][l15]      = (_Float16)(sA0[r] + bi0);
                slds[wv][qr][16 + l15] = (_Float16)(sA1[r] + bi1);
            }
            {
                int i = row0 + 16 + qr;
                int ti = i >> 8, hy = (i >> 4) & 15, wi = i & 15;
                float bi0 = btab[(((ti - tj0 + 3) * 31 + (hy - hj0 + 15)) * 31 + (wi - wj0 + 15)) * 8 + hd];
                float bi1 = btab[(((ti - tj1 + 3) * 31 + (hy - hj1 + 15)) * 31 + (wi - wj1 + 15)) * 8 + hd];
                slds[wv][16 + qr][l15]      = (_Float16)(sB0[r] + bi0);
                slds[wv][16 + qr][16 + l15] = (_Float16)(sB1[r] + bi1);
            }
        }
        __syncthreads();
        v16h saA, saB;
#pragma unroll
        for (int e = 0; e < 16; ++e) {
            int kk = kmap16(e, hi);
            saA[e] = slds[wv][l15][kk];                      // A: rows=q rows, K=m
            saB[e] = slds[wv][16 + l15][kk];
        }
        oA0 = __builtin_amdgcn_wmma_f32_16x16x32_f16(false, saA, false, vf0, (short)0, oA0, false, false);
        oA1 = __builtin_amdgcn_wmma_f32_16x16x32_f16(false, saA, false, vf1, (short)0, oA1, false, false);
        oB0 = __builtin_amdgcn_wmma_f32_16x16x32_f16(false, saB, false, vf0, (short)0, oB0, false, false);
        oB1 = __builtin_amdgcn_wmma_f32_16x16x32_f16(false, saB, false, vf1, (short)0, oB1, false, false);
        __syncthreads();
    }

    // o (b,h,tn,d) -> o_pre[t][b][c=h*32+d][n], scaled by 0.125
#pragma unroll
    for (int r = 0; r < 8; ++r) {
        int c0 = hd * 32 + l15, c1 = c0 + 16;
        {
            int i = row0 + r + 8 * hi;
            int t = i >> 8, n = i & 255;
            opre[((t * 8 + b) * 256 + c0) * 256 + n] = oA0[r] * 0.125f;
            opre[((t * 8 + b) * 256 + c1) * 256 + n] = oA1[r] * 0.125f;
        }
        {
            int i = row0 + 16 + r + 8 * hi;
            int t = i >> 8, n = i & 255;
            opre[((t * 8 + b) * 256 + c0) * 256 + n] = oB0[r] * 0.125f;
            opre[((t * 8 + b) * 256 + c1) * 256 + n] = oB1[r] * 0.125f;
        }
    }
}

// ---------------- LIF over T on attention output -> proj input spikes [B,C,TN] f16 ----------------
__global__ void lif_o_kernel(const float* __restrict__ opre, _Float16* __restrict__ pin) {
    int i = blockIdx.x * blockDim.x + threadIdx.x; // B*C*N
    if (i >= B_ * C_ * N_) return;
    int n = i & 255;
    int c = (i >> 8) & 255;
    int b = i >> 16;
    float mem = 0.f, sp = 0.f;
    for (int t = 0; t < T_; ++t) {
        float xv = opre[((t * 8 + b) * 256 + c) * 256 + n];
        mem = (mem - sp * 0.5f) * 0.25f + xv;
        sp = rintf(fminf(fmaxf(mem, 0.f), 1.f));
        pin[b * (C_ * TN_) + c * TN_ + t * 256 + n] = (_Float16)sp;
    }
}

// ---------------- Proj GEMM (256x256 x 256x1024 per b) + bias + BN ----------------
__global__ void proj_gemm_bn_kernel(const _Float16* __restrict__ wp, const _Float16* __restrict__ pin,
                                    const float* __restrict__ bp, const float* __restrict__ g,
                                    const float* __restrict__ be, const float* __restrict__ mu,
                                    const float* __restrict__ vr, float* __restrict__ pout) {
    int wave = (blockIdx.x * blockDim.x + threadIdx.x) >> 5;
    int lane = threadIdx.x & 31;
    int b   = wave / (16 * 16);
    int rem = wave % (16 * 16);
    int mt = rem / 16, nt = rem % 16;
    int l15 = lane & 15, hi = lane >> 4;

    v8f acc[4] = {};
    int mrow = mt * 16 + l15;
    const _Float16* pb = pin + b * (C_ * TN_) + l15 + nt * 64;
#pragma unroll
    for (int kc = 0; kc < 8; ++kc) {
        int K0 = kc * 32;
        v16h a;
#pragma unroll
        for (int e = 0; e < 16; ++e)
            a[e] = wp[mrow * 256 + K0 + kmap16(e, hi)];
        v16h bf[4];
#pragma unroll
        for (int j = 0; j < 4; ++j) {
#pragma unroll
            for (int e = 0; e < 16; ++e)
                bf[j][e] = pb[(K0 + kmap16(e, hi)) * TN_ + j * 16];
        }
#pragma unroll
        for (int j = 0; j < 4; ++j)
            acc[j] = __builtin_amdgcn_wmma_f32_16x16x32_f16(false, a, false, bf[j], (short)0, acc[j], false, false);
    }
#pragma unroll
    for (int j = 0; j < 4; ++j) {
        int ncol = nt * 64 + j * 16 + l15;
#pragma unroll
        for (int r = 0; r < 8; ++r) {
            int m = mt * 16 + r + 8 * hi;
            float y = acc[j][r] + bp[m];
            y = (y - mu[m]) * (g[m] * rsqrtf(vr[m] + 1e-5f)) + be[m];
            pout[b * (C_ * TN_) + m * TN_ + ncol] = y;
        }
    }
}

// ---------------- Final LIF -> output [T,B,C,H,W] f32 ----------------
__global__ void lif_final_kernel(const float* __restrict__ pout, float* __restrict__ out) {
    int i = blockIdx.x * blockDim.x + threadIdx.x; // B*C*N
    if (i >= B_ * C_ * N_) return;
    int n = i & 255;
    int c = (i >> 8) & 255;
    int b = i >> 16;
    float mem = 0.f, sp = 0.f;
    for (int t = 0; t < T_; ++t) {
        float xv = pout[b * (C_ * TN_) + c * TN_ + t * 256 + n];
        mem = (mem - sp * 0.5f) * 0.25f + xv;
        sp = rintf(fminf(fmaxf(mem, 0.f), 1.f));
        out[((t * 8 + b) * 256 + c) * 256 + n] = sp;
    }
}

extern "C" void kernel_launch(void* const* d_in, const int* in_sizes, int n_in,
                              void* d_out, int out_size, void* d_ws, size_t ws_size,
                              hipStream_t stream) {
    const float* x        = (const float*)d_in[0];
    const float* w_qkv    = (const float*)d_in[1];
    const float* qkv_g    = (const float*)d_in[2];
    const float* qkv_b    = (const float*)d_in[3];
    const float* qkv_m    = (const float*)d_in[4];
    const float* qkv_v    = (const float*)d_in[5];
    const float* btab     = (const float*)d_in[6];
    const float* w_proj   = (const float*)d_in[7];
    const float* b_proj   = (const float*)d_in[8];
    const float* proj_g   = (const float*)d_in[9];
    const float* proj_b   = (const float*)d_in[10];
    const float* proj_m   = (const float*)d_in[11];
    const float* proj_v   = (const float*)d_in[12];

    char* ws = (char*)d_ws;
    _Float16* xh      = (_Float16*)(ws + 0);            //  4,194,304 B
    _Float16* wqh     = (_Float16*)(ws + 4194304);      //    393,216 B
    _Float16* wph     = (_Float16*)(ws + 4587520);      //    131,072 B
    float*    qkv_pre = (float*)   (ws + 4718592);      // 25,165,824 B
    _Float16* q_h     = (_Float16*)(ws + 29884416);     //  4,194,304 B
    _Float16* k_h     = (_Float16*)(ws + 34078720);     //  4,194,304 B
    _Float16* v_h     = (_Float16*)(ws + 38273024);     //  4,194,304 B
    float*    o_pre   = (float*)   (ws + 42467328);     //  8,388,608 B
    _Float16* p_in    = (_Float16*)(ws + 50855936);     //  4,194,304 B
    float*    p_out   = (float*)   (ws + 55050240);     //  8,388,608 B

    cvt_f16_kernel<<<(2097152 + 255) / 256, 256, 0, stream>>>(x, xh, 2097152);
    cvt_f16_kernel<<<(196608 + 255) / 256, 256, 0, stream>>>(w_qkv, wqh, 196608);
    cvt_f16_kernel<<<(65536 + 255) / 256, 256, 0, stream>>>(w_proj, wph, 65536);

    qkv_gemm_bn_kernel<<<768, 256, 0, stream>>>(wqh, xh, qkv_g, qkv_b, qkv_m, qkv_v, qkv_pre);
    lif_qkv_kernel<<<(B_ * 768 * N_ + 255) / 256, 256, 0, stream>>>(qkv_pre, q_h, k_h, v_h);
    attn_kernel<<<512, 128, 0, stream>>>(q_h, k_h, v_h, btab, o_pre);
    lif_o_kernel<<<(B_ * C_ * N_ + 255) / 256, 256, 0, stream>>>(o_pre, p_in);
    proj_gemm_bn_kernel<<<256, 256, 0, stream>>>(wph, p_in, b_proj, proj_g, proj_b, proj_m, proj_v, p_out);
    lif_final_kernel<<<(B_ * C_ * N_ + 255) / 256, 256, 0, stream>>>(p_out, (float*)d_out);
}
